// FEMRTransformer_1984274891465
// MI455X (gfx1250) — matrix operand order repaired and checked by reading, not compiled, not run on hardware
//
#include <hip/hip_runtime.h>
#include <hip/hip_bf16.h>

#define H_SIZE   768
#define N_HEADS  12
#define HEADD    64
#define INTER    3072
#define N_LAYERS 6
#define LWIDTH   512
#define BATCH    4
#define SEQ      2048
#define TOKENS   (BATCH*SEQ)
#define QKV_N    (3*H_SIZE)      // 2304
#define COMB_N   (H_SIZE+INTER)  // 3840

typedef __bf16 bf16x16 __attribute__((ext_vector_type(16)));
typedef float  f32x8   __attribute__((ext_vector_type(8)));

union Frag { bf16x16 v; unsigned int u[8]; uint4 q[2]; };

__device__ __forceinline__ unsigned short f2bf(float f) {
  unsigned int u = __float_as_uint(f);
  u += 0x7FFFu + ((u >> 16) & 1u);          // round-to-nearest-even
  return (unsigned short)(u >> 16);
}

// CDNA5 async global->LDS copy (ASYNCcnt-tracked).  dst = LDS byte offset in a
// VGPR, src = 64-bit global address (VGPR pair).  GV addressing mode.
__device__ __forceinline__ void async_load_b128(unsigned lds_off, const void* gptr) {
  asm volatile("global_load_async_to_lds_b128 %0, %1, off"
               :: "v"(lds_off), "v"(gptr) : "memory");
}
#define WAIT_ASYNC(n) asm volatile("s_wait_asynccnt " #n ::: "memory")

__device__ __forceinline__ unsigned lds_off32(const void* p) {
  // flat pointers into the LDS aperture carry the workgroup-relative LDS
  // address in their low 32 bits (ISA 10.2 aperture rules)
  return (unsigned)(size_t)p;
}

// ---------------------------------------------------------------------------
// Weight convert + transpose: src f32 (L,K,N) -> dst bf16 (L,N,K)
// ---------------------------------------------------------------------------
__global__ void convert_transpose_kernel(const float* __restrict__ src,
                                         unsigned short* __restrict__ dst,
                                         int L, int K, int N) {
  size_t total = (size_t)L * K * N;
  for (size_t idx = (size_t)blockIdx.x * blockDim.x + threadIdx.x; idx < total;
       idx += (size_t)gridDim.x * blockDim.x) {
    int    k = (int)(idx % K);
    size_t r = idx / K;
    int    n = (int)(r % N);
    int    l = (int)(r / N);
    dst[idx] = f2bf(src[((size_t)l * K + k) * N + n]);
  }
}

// ---------------------------------------------------------------------------
// Block reduction over 768 elements handled by 256 threads (wave32)
// ---------------------------------------------------------------------------
__device__ __forceinline__ float block_sum_768(float v) {
  #pragma unroll
  for (int o = 16; o; o >>= 1) v += __shfl_down(v, o, 32);
  __shared__ float red[8];
  int lane = threadIdx.x & 31, wid = threadIdx.x >> 5;
  if (lane == 0) red[wid] = v;
  __syncthreads();
  float s = 0.0f;
  if (threadIdx.x < 8) s = red[threadIdx.x];
  if (threadIdx.x < 32) {
    s += __shfl_down(s, 4, 32);
    s += __shfl_down(s, 2, 32);
    s += __shfl_down(s, 1, 32);
  }
  if (threadIdx.x == 0) red[0] = s;
  __syncthreads();
  return red[0];
}

// ---------------------------------------------------------------------------
// x = rmsnorm(embed[token], in_norm_w)   (f32 out)
// ---------------------------------------------------------------------------
__global__ __launch_bounds__(256) void embed_rms_kernel(
    const int* __restrict__ tokens, const float* __restrict__ embed,
    const float* __restrict__ w, float* __restrict__ x) {
  int t = blockIdx.x, i = threadIdx.x;
  const float* e = embed + (size_t)tokens[t] * H_SIZE;
  float v0 = e[i], v1 = e[i + 256], v2 = e[i + 512];
  float ss = block_sum_768(v0 * v0 + v1 * v1 + v2 * v2);
  float inv = rsqrtf(ss * (1.0f / 768.0f) + 1e-6f);
  float* o = x + (size_t)t * H_SIZE;
  o[i] = v0 * inv * w[i];
  o[i + 256] = v1 * inv * w[i + 256];
  o[i + 512] = v2 * inv * w[i + 512];
}

// ---------------------------------------------------------------------------
// xn(bf16) = rmsnorm(x, norm_w)
// ---------------------------------------------------------------------------
__global__ __launch_bounds__(256) void rms_bf16_kernel(
    const float* __restrict__ x, const float* __restrict__ w,
    unsigned short* __restrict__ xn) {
  int t = blockIdx.x, i = threadIdx.x;
  const float* p = x + (size_t)t * H_SIZE;
  float v0 = p[i], v1 = p[i + 256], v2 = p[i + 512];
  float ss = block_sum_768(v0 * v0 + v1 * v1 + v2 * v2);
  float inv = rsqrtf(ss * (1.0f / 768.0f) + 1e-6f);
  unsigned short* o = xn + (size_t)t * H_SIZE;
  o[i] = f2bf(v0 * inv * w[i]);
  o[i + 256] = f2bf(v1 * inv * w[i + 256]);
  o[i + 512] = f2bf(v2 * inv * w[i + 512]);
}

// ---------------------------------------------------------------------------
// out(f32) = rmsnorm(x, out_norm_w)
// ---------------------------------------------------------------------------
__global__ __launch_bounds__(256) void final_rms_kernel(
    const float* __restrict__ x, const float* __restrict__ w,
    float* __restrict__ out) {
  int t = blockIdx.x, i = threadIdx.x;
  const float* p = x + (size_t)t * H_SIZE;
  float v0 = p[i], v1 = p[i + 256], v2 = p[i + 512];
  float ss = block_sum_768(v0 * v0 + v1 * v1 + v2 * v2);
  float inv = rsqrtf(ss * (1.0f / 768.0f) + 1e-6f);
  float* o = out + (size_t)t * H_SIZE;
  o[i] = v0 * inv * w[i];
  o[i + 256] = v1 * inv * w[i + 256];
  o[i + 512] = v2 * inv * w[i + 512];
}

// ---------------------------------------------------------------------------
// Tiled bf16 WMMA GEMM: C(MxN) = A(MxK) * Bt(NxK)^T + bias, fused epilogues.
// Block tile 128x128, 8 waves (2x4), wave tile 64x32 -> 8 v_wmma per K-slab.
// Double-buffered LDS staged by GLOBAL_LOAD_ASYNC_TO_LDS_B128 (ASYNCcnt).
// ---------------------------------------------------------------------------
enum { EPI_QKV = 0, EPI_FF = 1, EPI_OUT = 2 };

template <int EPI>
__global__ __launch_bounds__(256) void gemm_bf16_kernel(
    const unsigned short* __restrict__ A,   // M x K bf16 row-major
    const unsigned short* __restrict__ Bt,  // N x K bf16 row-major (pre-transposed)
    const float* __restrict__ bias,         // N
    const float* __restrict__ resid,        // M x 768 (EPI_OUT only)
    float* __restrict__ outF,
    unsigned short* __restrict__ outB,
    int K) {
  __shared__ unsigned int As[2][128 * 20];  // 128 rows x 16 u32 (+4 pad), x2 buf
  __shared__ unsigned int Bs[2][128 * 20];

  const int tid = threadIdx.x;
  const int lane = tid & 31;
  const int wid = tid >> 5;
  const int wm = wid >> 2;      // 0..1 : 64-row slice
  const int wn = wid & 3;       // 0..3 : 32-col slice
  const int g = lane >> 4;      // lane half
  const int l16 = lane & 15;
  const int mBase = blockIdx.y * 128;
  const int nBase = blockIdx.x * 128;

  f32x8 acc[4][2] = {};

  const int nk = K >> 5;
  const int ar0 = tid >> 2,         ac0 = tid & 3;          // rows 0..63
  const int ar1 = (tid + 256) >> 2, ac1 = (tid + 256) & 3;  // rows 64..127

  auto issue = [&](int buf, int k0) {  // 4 async b128 per thread -> ASYNCcnt+4/wave
    async_load_b128(lds_off32(&As[buf][ar0 * 20 + ac0 * 4]),
                    A + (size_t)(mBase + ar0) * K + k0 + ac0 * 8);
    async_load_b128(lds_off32(&As[buf][ar1 * 20 + ac1 * 4]),
                    A + (size_t)(mBase + ar1) * K + k0 + ac1 * 8);
    async_load_b128(lds_off32(&Bs[buf][ar0 * 20 + ac0 * 4]),
                    Bt + (size_t)(nBase + ar0) * K + k0 + ac0 * 8);
    async_load_b128(lds_off32(&Bs[buf][ar1 * 20 + ac1 * 4]),
                    Bt + (size_t)(nBase + ar1) * K + k0 + ac1 * 8);
  };

  issue(0, 0);
  for (int kt = 0; kt < nk; ++kt) {
    const int cur = kt & 1;
    if (kt + 1 < nk) {
      issue(cur ^ 1, (kt + 1) * 32);   // prefetch next slab into other buffer
      WAIT_ASYNC(4);                    // in-order: oldest 4 (current slab) done
    } else {
      WAIT_ASYNC(0);
    }
    __syncthreads();                    // every wave's slab landed in LDS

    Frag af[4], bfr[2];
    #pragma unroll
    for (int m = 0; m < 4; ++m) {
      int rr = wm * 64 + m * 16 + l16;
      af[m].q[0] = *(const uint4*)&As[cur][rr * 20 + g * 4];
      af[m].q[1] = *(const uint4*)&As[cur][rr * 20 + 8 + g * 4];
    }
    #pragma unroll
    for (int n = 0; n < 2; ++n) {
      int nn = wn * 32 + n * 16 + l16;
      bfr[n].q[0] = *(const uint4*)&Bs[cur][nn * 20 + g * 4];
      bfr[n].q[1] = *(const uint4*)&Bs[cur][nn * 20 + 8 + g * 4];
    }
    #pragma unroll
    for (int m = 0; m < 4; ++m)
      #pragma unroll
      for (int n = 0; n < 2; ++n)
        acc[m][n] = __builtin_amdgcn_wmma_f32_16x16x32_bf16(
            false, af[m].v, false, bfr[n].v, (short)0, acc[m][n], false, false);

    __syncthreads();  // all reads done before this buffer is refilled at kt+2
  }

  #pragma unroll
  for (int m = 0; m < 4; ++m)
    #pragma unroll
    for (int n = 0; n < 2; ++n)
      #pragma unroll
      for (int v = 0; v < 8; ++v) {
        int grow = mBase + wm * 64 + m * 16 + v + g * 8;
        int gcol = nBase + wn * 32 + n * 16 + l16;
        float val = acc[m][n][v] + bias[gcol];
        if (EPI == EPI_FF) {
          val = 0.5f * val * (1.0f + erff(val * 0.70710678118f));
          outB[(size_t)grow * COMB_N + H_SIZE + gcol] = f2bf(val);
        } else if (EPI == EPI_OUT) {
          val += resid[(size_t)grow * H_SIZE + gcol];
          outF[(size_t)grow * H_SIZE + gcol] = val;
        } else {
          outF[(size_t)grow * QKV_N + gcol] = val;
        }
      }
}

// ---------------------------------------------------------------------------
// RoPE: qkv f32 -> q,k bf16 (b,s,h,d) and v bf16 transposed (b,h,d,s)
// ---------------------------------------------------------------------------
__global__ __launch_bounds__(384) void rope_kernel(
    const float* __restrict__ qkv, const float* __restrict__ ages,
    unsigned short* __restrict__ qb, unsigned short* __restrict__ kb,
    unsigned short* __restrict__ vt) {
  int t = blockIdx.x;
  int tid = threadIdx.x;
  int h = tid >> 5, p = tid & 31;  // head, rotary pair
  int b = t >> 11, s = t & 2047;
  float invf = __expf(-(float)p * (2.0f / 31.0f) * 9.2103403719762f); // 10000^-lin
  float th = ages[t] * invf;
  float sn, cs;
  __sincosf(th, &sn, &cs);
  size_t base = (size_t)t * QKV_N + h * 64 + 2 * p;
  float q0 = qkv[base], q1 = qkv[base + 1];
  float k0 = qkv[base + H_SIZE], k1 = qkv[base + H_SIZE + 1];
  float v0 = qkv[base + 2 * H_SIZE], v1 = qkv[base + 2 * H_SIZE + 1];
  size_t ob = (size_t)t * H_SIZE + h * 64 + 2 * p;
  qb[ob] = f2bf(q0 * cs - q1 * sn);
  qb[ob + 1] = f2bf(q1 * cs + q0 * sn);
  kb[ob] = f2bf(k0 * cs - k1 * sn);
  kb[ob + 1] = f2bf(k1 * cs + k0 * sn);
  size_t vb = ((size_t)(b * N_HEADS + h) * 64 + 2 * p) * SEQ + s;
  vt[vb] = f2bf(v0);
  vt[vb + SEQ] = f2bf(v1);
}

// ---------------------------------------------------------------------------
// Flash-style causal local attention (window 512), one wave per 16-query tile
// per head.  Scores and PV both via v_wmma_f32_16x16x32_bf16; online softmax
// with 16-lane shfl_xor reductions; P re-laid out C->A through LDS.
// ---------------------------------------------------------------------------
__global__ __launch_bounds__(32) void attn_kernel(
    const unsigned short* __restrict__ qb, const unsigned short* __restrict__ kb,
    const unsigned short* __restrict__ vt, unsigned short* __restrict__ comb) {
  __shared__ unsigned int Pl[16 * 16];  // 16 rows x 32 bf16 probabilities
  const int lane = threadIdx.x;
  const int g = lane >> 4, l16 = lane & 15;
  const int qt = blockIdx.x & 127;
  const int h = (blockIdx.x >> 7) % N_HEADS;
  const int b = blockIdx.x / (128 * N_HEADS);
  const int q0 = qt * 16;
  const float NEGINF = -__builtin_inff();

  const unsigned int* qu = (const unsigned int*)qb;
  const unsigned int* ku = (const unsigned int*)kb;
  const unsigned int* vu = (const unsigned int*)vt;

  // Q A-fragments (d 0..31 and 32..63)
  Frag qa[2];
  {
    size_t base = (size_t)(b * SEQ + q0 + l16) * (H_SIZE / 2) + h * 32;
    #pragma unroll
    for (int f = 0; f < 2; ++f)
      #pragma unroll
      for (int i = 0; i < 4; ++i) {
        qa[f].u[i] = qu[base + f * 16 + g * 8 + i];
        qa[f].u[i + 4] = qu[base + f * 16 + g * 8 + 4 + i];
      }
  }

  f32x8 O[4] = {};
  float mrow[8], lrow[8];
  #pragma unroll
  for (int v = 0; v < 8; ++v) { mrow[v] = -1e30f; lrow[v] = 0.0f; }

  int jmin = q0 - (LWIDTH - 1);
  if (jmin < 0) jmin = 0;
  int jlo = jmin & ~31;

  for (int j0 = jlo; j0 <= q0; j0 += 32) {
    // ---- scores: two 16-key tiles, K-dim = 64 head dims (2 WMMAs each)
    f32x8 s0 = {}, s1 = {};
    #pragma unroll
    for (int nt = 0; nt < 2; ++nt) {
      int key = j0 + nt * 16 + l16;
      if (key > SEQ - 1) key = SEQ - 1;
      size_t kbase = (size_t)(b * SEQ + key) * (H_SIZE / 2) + h * 32;
      Frag kf0, kf1;
      #pragma unroll
      for (int i = 0; i < 8; ++i) {
        kf0.u[i] = ku[kbase + g * 8 + i];
        kf1.u[i] = ku[kbase + 16 + g * 8 + i];
      }
      f32x8 s = (nt == 0) ? s0 : s1;
      s = __builtin_amdgcn_wmma_f32_16x16x32_bf16(false, qa[0].v, false, kf0.v,
                                                  (short)0, s, false, false);
      s = __builtin_amdgcn_wmma_f32_16x16x32_bf16(false, qa[1].v, false, kf1.v,
                                                  (short)0, s, false, false);
      if (nt == 0) s0 = s; else s1 = s;
    }

    // ---- mask + online softmax (rows live across 16-lane halves)
    float mnew[8];
    #pragma unroll
    for (int v = 0; v < 8; ++v) {
      int qrow = q0 + v + g * 8;
      int key0 = j0 + l16, key1 = j0 + 16 + l16;
      float a = s0[v] * 0.125f;
      float c = s1[v] * 0.125f;
      if (!(key0 <= qrow && qrow - key0 < LWIDTH)) a = NEGINF;
      if (!(key1 <= qrow && qrow - key1 < LWIDTH)) c = NEGINF;
      s0[v] = a; s1[v] = c;
      float mx = fmaxf(a, c);
      mx = fmaxf(mx, __shfl_xor(mx, 1));
      mx = fmaxf(mx, __shfl_xor(mx, 2));
      mx = fmaxf(mx, __shfl_xor(mx, 4));
      mx = fmaxf(mx, __shfl_xor(mx, 8));
      mnew[v] = fmaxf(mrow[v], mx);
    }
    unsigned short* Ps = (unsigned short*)Pl;
    #pragma unroll
    for (int v = 0; v < 8; ++v) {
      float alpha = __expf(mrow[v] - mnew[v]);
      float p0 = __expf(s0[v] - mnew[v]);
      float p1 = __expf(s1[v] - mnew[v]);
      float rs = p0 + p1;
      rs += __shfl_xor(rs, 1);
      rs += __shfl_xor(rs, 2);
      rs += __shfl_xor(rs, 4);
      rs += __shfl_xor(rs, 8);
      lrow[v] = lrow[v] * alpha + rs;
      mrow[v] = mnew[v];
      O[0][v] *= alpha; O[1][v] *= alpha; O[2][v] *= alpha; O[3][v] *= alpha;
      Ps[(v + g * 8) * 32 + l16] = f2bf(p0);
      Ps[(v + g * 8) * 32 + 16 + l16] = f2bf(p1);
    }

    // ---- P as A-fragment (16 queries x 32 keys) from LDS
    Frag pf;
    #pragma unroll
    for (int i = 0; i < 4; ++i) {
      pf.u[i] = Pl[l16 * 16 + g * 8 + i];
      pf.u[i + 4] = Pl[l16 * 16 + g * 8 + 4 + i];
    }

    // ---- PV: 4 d-groups of 16, K-dim = 32 keys (contiguous dwords in vt,
    //      j0 <= q0 aligned down to 32 guarantees sp <= SEQ/2-1: no clamp so
    //      the 8 dwords coalesce into two b128 loads per fragment)
    #pragma unroll
    for (int dg = 0; dg < 4; ++dg) {
      Frag vf;
      size_t vbase = ((size_t)(b * N_HEADS + h) * 64 + dg * 16 + l16) * (SEQ / 2)
                   + (j0 >> 1) + g * 8;
      vf.q[0] = *(const uint4*)(vu + vbase);
      vf.q[1] = *(const uint4*)(vu + vbase + 4);
      O[dg] = __builtin_amdgcn_wmma_f32_16x16x32_bf16(false, pf.v, false, vf.v,
                                                      (short)0, O[dg], false, false);
    }
  }

  // ---- normalize + write into combined[:, 0:768] as bf16
  #pragma unroll
  for (int dg = 0; dg < 4; ++dg)
    #pragma unroll
    for (int v = 0; v < 8; ++v) {
      size_t row = (size_t)(b * SEQ + q0 + v + g * 8);
      comb[row * COMB_N + h * 64 + dg * 16 + l16] = f2bf(O[dg][v] / lrow[v]);
    }
}

// ---------------------------------------------------------------------------
// Launcher
// ---------------------------------------------------------------------------
extern "C" void kernel_launch(void* const* d_in, const int* in_sizes, int n_in,
                              void* d_out, int out_size, void* d_ws, size_t ws_size,
                              hipStream_t stream) {
  (void)in_sizes; (void)n_in; (void)out_size; (void)ws_size;
  const int*   tokens   = (const int*)d_in[0];
  const float* ages     = (const float*)d_in[1];
  const float* embed    = (const float*)d_in[2];
  const float* in_norm  = (const float*)d_in[3];
  const float* out_norm = (const float*)d_in[4];
  const float* norm_w   = (const float*)d_in[5];
  const float* attn_W   = (const float*)d_in[6];
  const float* attn_b   = (const float*)d_in[7];
  const float* in_W     = (const float*)d_in[8];
  const float* in_b     = (const float*)d_in[9];
  const float* out_W    = (const float*)d_in[10];
  const float* out_b    = (const float*)d_in[11];

  char* ws = (char*)d_ws;
  size_t off = 0;
  auto take = [&](size_t bytes) -> char* {
    char* p = ws + off;
    off += (bytes + 255) & ~(size_t)255;
    return p;
  };
  unsigned short* wAttn = (unsigned short*)take((size_t)N_LAYERS * QKV_N * H_SIZE * 2);
  unsigned short* wIn   = (unsigned short*)take((size_t)N_LAYERS * INTER * H_SIZE * 2);
  unsigned short* wOut  = (unsigned short*)take((size_t)N_LAYERS * H_SIZE * COMB_N * 2);
  float*          x     = (float*)take((size_t)TOKENS * H_SIZE * 4);
  unsigned short* xn    = (unsigned short*)take((size_t)TOKENS * H_SIZE * 2);
  float*          qkv   = (float*)take((size_t)TOKENS * QKV_N * 4);
  unsigned short* qb    = (unsigned short*)take((size_t)TOKENS * H_SIZE * 2);
  unsigned short* kb    = (unsigned short*)take((size_t)TOKENS * H_SIZE * 2);
  unsigned short* vt    = (unsigned short*)take((size_t)TOKENS * H_SIZE * 2);
  unsigned short* comb  = (unsigned short*)take((size_t)TOKENS * COMB_N * 2);

  convert_transpose_kernel<<<8192, 256, 0, stream>>>(attn_W, wAttn, N_LAYERS, H_SIZE, QKV_N);
  convert_transpose_kernel<<<8192, 256, 0, stream>>>(in_W,  wIn,  N_LAYERS, H_SIZE, INTER);
  convert_transpose_kernel<<<8192, 256, 0, stream>>>(out_W, wOut, N_LAYERS, COMB_N, H_SIZE);

  embed_rms_kernel<<<TOKENS, 256, 0, stream>>>(tokens, embed, in_norm, x);

  for (int l = 0; l < N_LAYERS; ++l) {
    rms_bf16_kernel<<<TOKENS, 256, 0, stream>>>(x, norm_w + l * H_SIZE, xn);

    gemm_bf16_kernel<EPI_QKV><<<dim3(QKV_N / 128, TOKENS / 128), 256, 0, stream>>>(
        xn, wAttn + (size_t)l * QKV_N * H_SIZE, attn_b + l * QKV_N,
        nullptr, qkv, nullptr, H_SIZE);

    rope_kernel<<<TOKENS, 384, 0, stream>>>(qkv, ages, qb, kb, vt);

    attn_kernel<<<BATCH * N_HEADS * (SEQ / 16), 32, 0, stream>>>(qb, kb, vt, comb);

    gemm_bf16_kernel<EPI_FF><<<dim3(INTER / 128, TOKENS / 128), 256, 0, stream>>>(
        xn, wIn + (size_t)l * INTER * H_SIZE, in_b + l * INTER,
        nullptr, nullptr, comb, H_SIZE);

    gemm_bf16_kernel<EPI_OUT><<<dim3(H_SIZE / 128, TOKENS / 128), 256, 0, stream>>>(
        comb, wOut + (size_t)l * H_SIZE * COMB_N, out_b + l * H_SIZE,
        x, x, nullptr, COMB_N);
  }

  final_rms_kernel<<<TOKENS, 256, 0, stream>>>(x, out_norm, (float*)d_out);
}
